// TransformerEncoder_49984829391400
// MI455X (gfx1250) — matrix operand order repaired and checked by reading, not compiled, not run on hardware
//
#include <hip/hip_runtime.h>

#define BB 16
#define LL 256
#define DD 1024
#define FF 4096
#define HH 16
#define HD 64
#define NLAYER 4
#define L2X 512   // 2*L

typedef __attribute__((ext_vector_type(16))) __bf16 v16bf;
typedef __attribute__((ext_vector_type(8)))  float v8f;
typedef __attribute__((ext_vector_type(4)))  unsigned int v4u;
typedef __attribute__((ext_vector_type(4)))  unsigned int u32x4;
typedef __attribute__((ext_vector_type(8)))  int i32x8;
typedef __attribute__((ext_vector_type(4)))  int i32x4;

// Tensor Data Mover availability (guarded; falls back to vector staging)
#if defined(__has_builtin)
#if __has_builtin(__builtin_amdgcn_tensor_load_to_lds) && __has_builtin(__builtin_amdgcn_s_wait_tensorcnt)
#define USE_TDM 1
#else
#define USE_TDM 0
#endif
#else
#define USE_TDM 0
#endif

__device__ __forceinline__ __bf16 f2bf(float f) {
  unsigned u = __builtin_bit_cast(unsigned, f);
  u += 0x7FFFu + ((u >> 16) & 1u);            // round to nearest even
  unsigned short h = (unsigned short)(u >> 16);
  return __builtin_bit_cast(__bf16, h);
}
__device__ __forceinline__ float bf2f(__bf16 b) {
  unsigned u = ((unsigned)__builtin_bit_cast(unsigned short, b)) << 16;
  return __builtin_bit_cast(float, u);
}

enum { EPI_F32 = 0, EPI_BF16 = 1, EPI_BF16_BIAS_RELU = 2, EPI_F32_BIAS = 3 };

#if USE_TDM
// TDM 2-D tile load: gptr -> LDS byte offset ldsOff; tile = rows x 32 bf16,
// row stride = ld elements. D# layout per CDNA5 ISA 8.3/8.4.
__device__ __forceinline__ void tdm_load_tile(const __bf16* gptr, unsigned ldsOff,
                                              int rows, int ld) {
  unsigned long long ga = (unsigned long long)gptr;
  u32x4 g0;
  g0.x = 1u;                                            // count=1, user descriptor
  g0.y = ldsOff;                                        // lds_addr
  g0.z = (unsigned)(ga & 0xFFFFFFFFull);                // global_addr[31:0]
  g0.w = (unsigned)((ga >> 32) & 0x01FFFFFFull) | (2u << 30);  // addr[56:32] | type=2
  i32x8 g1;
  g1[0] = (1 << 16);                                    // wg_mask=0, data_size=1 (2B)
  g1[1] = (32 & 0xFFFF) << 16;                          // tensor_dim0[15:0]=32 (tile bound)
  g1[2] = 0 | ((rows & 0xFFFF) << 16);                  // tensor_dim0[31:16] | tensor_dim1[15:0]
  g1[3] = ((rows >> 16) & 0xFFFF) | (32 << 16);         // tensor_dim1[31:16] | tile_dim0=32
  g1[4] = rows & 0xFFFF;                                // tile_dim1=rows, tile_dim2=0
  g1[5] = ld;                                           // tensor_dim0_stride[31:0]
  g1[6] = 0;                                            // stride[47:32]=0 | dim1_stride[15:0]=0
  g1[7] = 0;
  i32x4 z4 = {0, 0, 0, 0};
  i32x8 z8 = {0, 0, 0, 0, 0, 0, 0, 0};
  (void)z8;
#if __clang_major__ >= 23
  __builtin_amdgcn_tensor_load_to_lds(g0, g1, z4, z4, z8, 0);
#else
  __builtin_amdgcn_tensor_load_to_lds(g0, g1, z4, z4, 0);
#endif
}
#endif

// ---------------------------------------------------------------------------
// Tiled bf16 WMMA GEMM: C = A(MxK, row-major) @ Bt^T, where Bt is B transposed
// (N x K row-major). Both operands are contiguous row panels -> both staged by
// the Tensor Data Mover into double-buffered LDS, pipelined against the WMMAs.
// 256 threads = 8 waves (wave32). Block tile BM x BN, K-step 32.
// LDS: 2 x [A tile BM*32 bf16 | B tile BN*32 bf16].
// ---------------------------------------------------------------------------
template <int BM, int BN, int WM_WAVES, int WN_WAVES, int EPI>
__global__ __launch_bounds__(256)
void gemm_bf16(const __bf16* __restrict__ A, const __bf16* __restrict__ Bt,
               void* __restrict__ Cv, const float* __restrict__ bias,
               int M, int N, int K,
               long long strideA, long long strideB, long long strideC,
               int lda, int ldb, int ldc)
{
  (void)M; (void)N;
  constexpr int BK = 32;
  constexpr int TM = BM / (16 * WM_WAVES);
  constexpr int TN = BN / (16 * WN_WAVES);
  constexpr unsigned ABYTES = BM * BK * 2;
  constexpr unsigned BUFBYTES = (BM + BN) * BK * 2;
  extern __shared__ char smem[];

  const int tid  = threadIdx.x;
  const int wave = tid >> 5;
  const int lane = tid & 31;
  const int l16  = lane & 15;
  const int lhalf = lane >> 4;
  const int kbA  = lhalf * 8;     // A frag: K = kbA..kbA+7 and kbA+16..kbA+23
  const int kbB  = lhalf * 8;     // B frag: K = kbB..kbB+7 and kbB+16..kbB+23
  const int wm   = wave / WN_WAVES;
  const int wn   = wave % WN_WAVES;
  const int bm0  = blockIdx.y * BM;
  const int bn0  = blockIdx.x * BN;

  A  += (long long)blockIdx.z * strideA;
  Bt += (long long)blockIdx.z * strideB;

  union Frag { v16bf v; v4u u[2]; };

  v8f acc[TM][TN] = {};
  const int nt = K / BK;

  auto stage = [&](int t, int buf) {
#if USE_TDM
    if (wave == 0) {
      tdm_load_tile(&A [(size_t)bm0 * lda + t * BK], buf * BUFBYTES,          BM, lda);
      tdm_load_tile(&Bt[(size_t)bn0 * ldb + t * BK], buf * BUFBYTES + ABYTES, BN, ldb);
    }
#else
    __bf16 (*As)[BK] = (__bf16(*)[BK])(smem + buf * BUFBYTES);
    __bf16 (*Bs)[BK] = (__bf16(*)[BK])(smem + buf * BUFBYTES + ABYTES);
    constexpr int A_ITERS = (BM * (BK / 8)) / 256;
    #pragma unroll
    for (int it = 0; it < A_ITERS; ++it) {
      int i = tid + it * 256;
      int r = i >> 2, c = (i & 3) * 8;
      *(v4u*)&As[r][c] = *(const v4u*)&A[(size_t)(bm0 + r) * lda + t * BK + c];
    }
    constexpr int B_ITERS = (BN * (BK / 8)) / 256;
    #pragma unroll
    for (int it = 0; it < B_ITERS; ++it) {
      int i = tid + it * 256;
      int r = i >> 2, c = (i & 3) * 8;
      *(v4u*)&Bs[r][c] = *(const v4u*)&Bt[(size_t)(bn0 + r) * ldb + t * BK + c];
    }
#endif
  };

  stage(0, 0);
  for (int t = 0; t < nt; ++t) {
    const int cur = t & 1;
    if (t + 1 < nt) stage(t + 1, cur ^ 1);     // DMA next tile while we compute
#if USE_TDM
    if (wave == 0) {
      if (t + 1 < nt) __builtin_amdgcn_s_wait_tensorcnt(2);  // tile t retired (in-order)
      else            __builtin_amdgcn_s_wait_tensorcnt(0);
    }
#endif
    __syncthreads();

    __bf16 (*As)[BK] = (__bf16(*)[BK])(smem + cur * BUFBYTES);
    __bf16 (*Bs)[BK] = (__bf16(*)[BK])(smem + cur * BUFBYTES + ABYTES);

    Frag bfr[TN];
    #pragma unroll
    for (int tn = 0; tn < TN; ++tn) {
      int col = wn * (TN * 16) + tn * 16 + l16;
      bfr[tn].u[0] = *(const v4u*)&Bs[col][kbB];
      bfr[tn].u[1] = *(const v4u*)&Bs[col][kbB + 16];
    }
    #pragma unroll
    for (int tm = 0; tm < TM; ++tm) {
      Frag af;
      int row = wm * (TM * 16) + tm * 16 + l16;
      af.u[0] = *(const v4u*)&As[row][kbA];
      af.u[1] = *(const v4u*)&As[row][kbA + 16];
      #pragma unroll
      for (int tn = 0; tn < TN; ++tn) {
        acc[tm][tn] = __builtin_amdgcn_wmma_f32_16x16x32_bf16(
            false, af.v, false, bfr[tn].v, (short)0, acc[tm][tn], false, false);
      }
    }
    __syncthreads();   // protect buf (cur) from being overwritten by stage(t+2)
  }

  float*  Cf = (float*)Cv  + (long long)blockIdx.z * strideC;
  __bf16* Ch = (__bf16*)Cv + (long long)blockIdx.z * strideC;
  const int crow0 = bm0 + wm * (TM * 16) + lhalf * 8;
  const int ccol0 = bn0 + wn * (TN * 16) + l16;
  #pragma unroll
  for (int tm = 0; tm < TM; ++tm)
    #pragma unroll
    for (int tn = 0; tn < TN; ++tn)
      #pragma unroll
      for (int i = 0; i < 8; ++i) {
        int r = crow0 + tm * 16 + i;
        int c = ccol0 + tn * 16;
        float val = acc[tm][tn][i];
        if constexpr (EPI == EPI_BF16_BIAS_RELU) {
          val += bias[c]; val = fmaxf(val, 0.0f);
          Ch[(size_t)r * ldc + c] = f2bf(val);
        } else if constexpr (EPI == EPI_F32_BIAS) {
          val += bias[c];
          Cf[(size_t)r * ldc + c] = val;
        } else if constexpr (EPI == EPI_BF16) {
          Ch[(size_t)r * ldc + c] = f2bf(val);
        } else {
          Cf[(size_t)r * ldc + c] = val;
        }
      }
}

// ---------------------------------------------------------------------------
// Sinusoidal relative position table: pos[p][j], p in [0,512), n = p+1
// ---------------------------------------------------------------------------
__global__ void pos_kernel(float* __restrict__ pos) {
  int idx = blockIdx.x * 256 + threadIdx.x;      // 512*64
  int p = idx >> 6, j = idx & 63;
  int jj = j & 31;
  float freq = expf((float)jj * (-9.210340371976184f / 31.0f));
  float a = (float)(p + 1) * freq;
  pos[idx] = (j < 32) ? sinf(a) : cosf(a);
}

// r = pos @ Wr, natural layout rn[p][j] (2L x hd) == Bt for the BD GEMM
__global__ void compute_rt(const float* __restrict__ pos, const float* __restrict__ Wr,
                           __bf16* __restrict__ rn) {
  int idx = blockIdx.x * 256 + threadIdx.x;      // 512*64
  int p = idx >> 6;
  int j = idx & 63;
  float acc = 0.0f;
  #pragma unroll
  for (int k = 0; k < HD; ++k) acc += pos[p * HD + k] * Wr[k * HD + j];
  rn[idx] = f2bf(acc);
}

__global__ void init_h(const float* __restrict__ h, float* __restrict__ hcur,
                       __bf16* __restrict__ hb) {
  size_t idx = (size_t)blockIdx.x * 256 + threadIdx.x;
  float v = h[idx];
  hcur[idx] = v;
  hb[idx] = f2bf(v);
}

// f32 (R x C) -> bf16 transposed (C x R): weights become Bt panels once per layer
__global__ void cvt_bf16_t(const float* __restrict__ in, __bf16* __restrict__ out,
                           int R, int C) {
  int idx = blockIdx.x * 256 + threadIdx.x;
  int r = idx / C, c = idx % C;
  out[(size_t)c * R + r] = f2bf(in[idx]);
}

// q/kv f32 -> bf16 head-major layouts with rel-pos biases folded in.
// kn = K natural (Bt for AC); vt = V transposed (Bt for PV).
__global__ void attn_prep(const float* __restrict__ q, const float* __restrict__ kv,
                          const float* __restrict__ rrb, const float* __restrict__ rwb,
                          __bf16* __restrict__ q1, __bf16* __restrict__ q2,
                          __bf16* __restrict__ kn, __bf16* __restrict__ vt) {
  size_t idx = (size_t)blockIdx.x * 256 + threadIdx.x;   // over B*L*D
  int c = (int)(idx & (DD - 1));
  size_t bl = idx >> 10;
  int l = (int)(bl & (LL - 1));
  int b = (int)(bl >> 8);
  int h = c >> 6, t = c & 63;
  size_t bh = (size_t)b * HH + h;
  float qq = q[idx];
  size_t qidx = (bh * LL + l) * HD + t;
  q1[qidx] = f2bf(qq + rrb[h * HD + t]);
  q2[qidx] = f2bf(qq + rwb[h * HD + t]);
  kn[qidx] = f2bf(kv[bl * (2 * DD) + c]);                        // K natural
  vt[(bh * HD + t) * LL + l] = f2bf(kv[bl * (2 * DD) + DD + c]); // V transposed
}

// Fused relative-shift + mask + softmax: one block (256 thr) per query row
__global__ void attn_softmax(const __bf16* __restrict__ AC, const __bf16* __restrict__ BD,
                             const int* __restrict__ mask, __bf16* __restrict__ P) {
  __shared__ float red[256];
  int row = blockIdx.x;            // bh*L + i
  int i  = row & (LL - 1);
  int bh = row >> 8;
  int b  = bh >> 4;
  int j  = threadIdx.x;
  float s = (bf2f(AC[(size_t)row * LL + j]) +
             bf2f(BD[(size_t)bh * LL * L2X + (size_t)i * L2X + (j - i + LL)])) * 0.125f;
  if (mask[b * LL + j] == 0) s = -3.0e38f;
  red[j] = s; __syncthreads();
  for (int st = 128; st > 0; st >>= 1) { if (j < st) red[j] = fmaxf(red[j], red[j + st]); __syncthreads(); }
  float m = red[0]; __syncthreads();
  float e = expf(s - m);
  red[j] = e; __syncthreads();
  for (int st = 128; st > 0; st >>= 1) { if (j < st) red[j] += red[j + st]; __syncthreads(); }
  float inv = 1.0f / red[0];
  P[(size_t)row * LL + j] = f2bf(e * inv);
}

// Residual add + LayerNorm; ATTN selects head-major input layout for x1.
template <bool ATTN>
__global__ void add_layernorm(const float* __restrict__ x1, const float* __restrict__ res,
                              const float* __restrict__ g, const float* __restrict__ bb,
                              float* __restrict__ hout, __bf16* __restrict__ hb) {
  __shared__ float red[256];
  int row = blockIdx.x;            // b*L + l
  int b = row >> 8, l = row & (LL - 1);
  float v[4];
  float s = 0.0f;
  #pragma unroll
  for (int it = 0; it < 4; ++it) {
    int c = threadIdx.x + it * 256;
    float xv;
    if constexpr (ATTN) {
      int h = c >> 6, t = c & 63;
      xv = x1[(((size_t)b * HH + h) * LL + l) * HD + t];
    } else {
      xv = x1[(size_t)row * DD + c];
    }
    xv += res[(size_t)row * DD + c];
    v[it] = xv; s += xv;
  }
  red[threadIdx.x] = s; __syncthreads();
  for (int st = 128; st > 0; st >>= 1) { if (threadIdx.x < st) red[threadIdx.x] += red[threadIdx.x + st]; __syncthreads(); }
  float mean = red[0] * (1.0f / (float)DD);
  __syncthreads();
  s = 0.0f;
  #pragma unroll
  for (int it = 0; it < 4; ++it) { float d = v[it] - mean; s += d * d; }
  red[threadIdx.x] = s; __syncthreads();
  for (int st = 128; st > 0; st >>= 1) { if (threadIdx.x < st) red[threadIdx.x] += red[threadIdx.x + st]; __syncthreads(); }
  float inv = rsqrtf(red[0] * (1.0f / (float)DD) + 1e-12f);
  #pragma unroll
  for (int it = 0; it < 4; ++it) {
    int c = threadIdx.x + it * 256;
    float y = (v[it] - mean) * inv * g[c] + bb[c];
    hout[(size_t)row * DD + c] = y;
    hb[(size_t)row * DD + c] = f2bf(y);
  }
}

// ---------------------------------------------------------------------------
extern "C" void kernel_launch(void* const* d_in, const int* in_sizes, int n_in,
                              void* d_out, int out_size, void* d_ws, size_t ws_size,
                              hipStream_t stream) {
  (void)in_sizes; (void)n_in; (void)out_size; (void)ws_size;
  const float* h_in = (const float*)d_in[0];
  const int*   mask = (const int*)d_in[1];
  const float* Wq   = (const float*)d_in[2];
  const float* Wkv  = (const float*)d_in[3];
  const float* Wr   = (const float*)d_in[4];
  const float* rrb  = (const float*)d_in[5];
  const float* rwb  = (const float*)d_in[6];
  const float* ln1g = (const float*)d_in[7];
  const float* ln1b = (const float*)d_in[8];
  const float* ln2g = (const float*)d_in[9];
  const float* ln2b = (const float*)d_in[10];
  const float* W1   = (const float*)d_in[11];
  const float* b1   = (const float*)d_in[12];
  const float* W2   = (const float*)d_in[13];
  const float* b2   = (const float*)d_in[14];

  char* base = (char*)d_ws;
  size_t off = 0;
  auto take = [&](size_t bytes) -> char* {
    char* p = base + off;
    off += (bytes + 255) & ~(size_t)255;
    return p;
  };
  float*  posb = (float*) take((size_t)L2X * HD * 4);
  float*  hcur = (float*) take((size_t)BB * LL * DD * 4);
  __bf16* hb   = (__bf16*)take((size_t)BB * LL * DD * 2);
  float*  qf   = (float*) take((size_t)BB * LL * DD * 4);
  float*  kvf  = (float*) take((size_t)BB * LL * 2 * DD * 4);
  __bf16* q1   = (__bf16*)take((size_t)BB * HH * LL * HD * 2);
  __bf16* q2   = (__bf16*)take((size_t)BB * HH * LL * HD * 2);
  __bf16* kn   = (__bf16*)take((size_t)BB * HH * LL * HD * 2);
  __bf16* vt   = (__bf16*)take((size_t)BB * HH * HD * LL * 2);
  __bf16* rn   = (__bf16*)take((size_t)L2X * HD * 2);
  __bf16* ac   = (__bf16*)take((size_t)BB * HH * LL * LL * 2);
  __bf16* bd   = (__bf16*)take((size_t)BB * HH * LL * L2X * 2);
  __bf16* pm   = (__bf16*)take((size_t)BB * HH * LL * LL * 2);
  float*  aout = (float*) take((size_t)BB * HH * LL * HD * 4);
  __bf16* wqt  = (__bf16*)take((size_t)DD * DD * 2);        // Wq^T  (DD x DD)
  __bf16* wkvt = (__bf16*)take((size_t)DD * 2 * DD * 2);    // Wkv^T (2DD x DD)
  __bf16* w1t  = (__bf16*)take((size_t)DD * FF * 2);        // W1^T  (FF x DD)
  __bf16* w2t  = (__bf16*)take((size_t)FF * DD * 2);        // W2^T  (DD x FF)
  __bf16* ffa  = (__bf16*)take((size_t)BB * LL * FF * 2);
  float*  ffo  = (float*) take((size_t)BB * LL * DD * 4);

  constexpr size_t LDS_128x128 = 2 * (size_t)(128 + 128) * 32 * 2;   // 32 KB
  constexpr size_t LDS_128x64  = 2 * (size_t)(128 + 64) * 32 * 2;    // 24 KB

  pos_kernel<<<dim3((L2X * HD) / 256), 256, 0, stream>>>(posb);
  init_h<<<dim3((BB * LL * DD) / 256), 256, 0, stream>>>(h_in, hcur, hb);

  for (int i = 0; i < NLAYER; ++i) {
    cvt_bf16_t<<<dim3((DD * DD) / 256), 256, 0, stream>>>(Wq + (size_t)i * DD * DD, wqt, DD, DD);
    cvt_bf16_t<<<dim3((DD * 2 * DD) / 256), 256, 0, stream>>>(Wkv + (size_t)i * DD * 2 * DD, wkvt, DD, 2 * DD);
    cvt_bf16_t<<<dim3((DD * FF) / 256), 256, 0, stream>>>(W1 + (size_t)i * DD * FF, w1t, DD, FF);
    cvt_bf16_t<<<dim3((FF * DD) / 256), 256, 0, stream>>>(W2 + (size_t)i * FF * DD, w2t, FF, DD);
    compute_rt<<<dim3((HD * L2X) / 256), 256, 0, stream>>>(posb, Wr + (size_t)i * HD * HD, rn);

    // q = h @ Wq  (4096x1024x1024), Bt = Wq^T (ldb = DD)
    gemm_bf16<128, 128, 2, 4, EPI_F32><<<dim3(DD / 128, (BB * LL) / 128, 1), 256, LDS_128x128, stream>>>(
        hb, wqt, qf, nullptr, BB * LL, DD, DD, 0, 0, 0, DD, DD, DD);
    // kv = h @ Wkv (4096x2048x1024), Bt = Wkv^T (ldb = DD)
    gemm_bf16<128, 128, 2, 4, EPI_F32><<<dim3((2 * DD) / 128, (BB * LL) / 128, 1), 256, LDS_128x128, stream>>>(
        hb, wkvt, kvf, nullptr, BB * LL, 2 * DD, DD, 0, 0, 0, DD, DD, 2 * DD);

    attn_prep<<<dim3((BB * LL * DD) / 256), 256, 0, stream>>>(
        qf, kvf, rrb + (size_t)i * HH * HD, rwb + (size_t)i * HH * HD, q1, q2, kn, vt);

    // AC = (q+rrb) @ K^T, batched over 256 (b,h); Bt = K natural (ldb = HD)
    gemm_bf16<128, 128, 2, 4, EPI_BF16><<<dim3(LL / 128, LL / 128, BB * HH), 256, LDS_128x128, stream>>>(
        q1, kn, ac, nullptr, LL, LL, HD,
        (long long)LL * HD, (long long)LL * HD, (long long)LL * LL, HD, HD, LL);
    // BD = (q+rwb) @ R^T, batched; Bt = R natural (ldb = HD, strideB = 0)
    gemm_bf16<128, 128, 2, 4, EPI_BF16><<<dim3(L2X / 128, LL / 128, BB * HH), 256, LDS_128x128, stream>>>(
        q2, rn, bd, nullptr, LL, L2X, HD,
        (long long)LL * HD, 0LL, (long long)LL * L2X, HD, HD, L2X);

    attn_softmax<<<dim3(BB * HH * LL), 256, 0, stream>>>(ac, bd, mask, pm);

    // out = P @ V, batched (256x64x256); Bt = V^T (ldb = LL)
    gemm_bf16<128, 64, 4, 2, EPI_F32><<<dim3(HD / 64, LL / 128, BB * HH), 256, LDS_128x64, stream>>>(
        pm, vt, aout, nullptr, LL, HD, LL,
        (long long)LL * LL, (long long)HD * LL, (long long)LL * HD, LL, LL, HD);

    add_layernorm<true><<<dim3(BB * LL), 256, 0, stream>>>(
        aout, hcur, ln1g + (size_t)i * DD, ln1b + (size_t)i * DD, hcur, hb);

    // FFN; Bt = W1^T (ldb = DD), W2^T (ldb = FF)
    gemm_bf16<128, 128, 2, 4, EPI_BF16_BIAS_RELU><<<dim3(FF / 128, (BB * LL) / 128, 1), 256, LDS_128x128, stream>>>(
        hb, w1t, ffa, b1 + (size_t)i * FF, BB * LL, FF, DD, 0, 0, 0, DD, DD, FF);
    gemm_bf16<128, 128, 2, 4, EPI_F32_BIAS><<<dim3(DD / 128, (BB * LL) / 128, 1), 256, LDS_128x128, stream>>>(
        ffa, w2t, ffo, b2 + (size_t)i * DD, BB * LL, DD, FF, 0, 0, 0, FF, FF, DD);

    add_layernorm<false><<<dim3(BB * LL), 256, 0, stream>>>(
        ffo, hcur, ln2g + (size_t)i * DD, ln2b + (size_t)i * DD, hcur, hb);
  }

  hipMemcpyAsync(d_out, hcur, (size_t)BB * LL * DD * 4, hipMemcpyDeviceToDevice, stream);
}
